// LSTM_53188874993812
// MI455X (gfx1250) — matrix operand order repaired and checked by reading, not compiled, not run on hardware
//
#include <hip/hip_runtime.h>
#include <hip/hip_bf16.h>
#include <math.h>
#include <stdint.h>

typedef _Float16 half_t;
typedef __attribute__((ext_vector_type(16))) _Float16 v16h;
typedef __attribute__((ext_vector_type(8)))  _Float16 h8;
typedef __attribute__((ext_vector_type(8)))  float    v8f;

#define B_DIM  8192
#define H_DIM  1024
#define KCAT   2048   // IN + H
#define NG     4096   // 4 * H  (f,i,c,o concatenated)
#define BHE    (B_DIM * H_DIM)

// Use GLOBAL_LOAD_ASYNC_TO_LDS_B128 (CDNA5 async DMA, ASYNCcnt) for staging.
#define STAGE_ASYNC 1

// ---------------------------------------------------------------------------
// Pack kernels: one-time f32 -> f16 conversion + layout for WMMA.
// ---------------------------------------------------------------------------
__global__ void pack_A(const float* __restrict__ inp, const float* __restrict__ hid,
                       half_t* __restrict__ Abuf) {
    long idx = (long)blockIdx.x * blockDim.x + threadIdx.x;   // B*2048 threads
    long b = idx >> 11; int k = (int)(idx & 2047);
    float v = (k < H_DIM) ? inp[b * H_DIM + k] : hid[b * H_DIM + (k - H_DIM)];
    Abuf[idx] = (half_t)v;
}

__global__ void pack_WUT(const float* __restrict__ Wf, const float* __restrict__ Wi,
                         const float* __restrict__ Wc, const float* __restrict__ Wo,
                         const float* __restrict__ Uf, const float* __restrict__ Ui,
                         const float* __restrict__ Uc, const float* __restrict__ Uo,
                         half_t* __restrict__ WUT) {
    long idx = (long)blockIdx.x * blockDim.x + threadIdx.x;   // 4096*2048 threads
    int n = (int)(idx >> 11); int k = (int)(idx & 2047);
    int g = n >> 10; int h = n & 1023;
    const float* W; const float* U;
    switch (g) {
        case 0:  W = Wf; U = Uf; break;
        case 1:  W = Wi; U = Ui; break;
        case 2:  W = Wc; U = Uc; break;
        default: W = Wo; U = Uo; break;
    }
    float v = (k < H_DIM) ? W[(long)k * H_DIM + h] : U[(long)(k - H_DIM) * H_DIM + h];
    WUT[idx] = (half_t)v;
}

__global__ void pack_bcat(const float* __restrict__ bf, const float* __restrict__ bi,
                          const float* __restrict__ bc, const float* __restrict__ bo,
                          float* __restrict__ bcat) {
    int n = blockIdx.x * blockDim.x + threadIdx.x;            // 4096 threads
    const float* src = (n < 1024) ? bf : (n < 2048) ? bi : (n < 3072) ? bc : bo;
    bcat[n] = src[n & 1023];
}

__global__ void pack_WhT(const float* __restrict__ Wh, half_t* __restrict__ WhT) {
    long idx = (long)blockIdx.x * blockDim.x + threadIdx.x;   // 1024*1024 threads
    int n = (int)(idx >> 10); int k = (int)(idx & 1023);
    WhT[idx] = (half_t)Wh[(long)k * H_DIM + n];
}

// ---------------------------------------------------------------------------
// WMMA GEMM:  C[M,N] = A[M,K] * Bt[N,K]^T + bias[N]   (f16 in, f32 accum)
// 256 threads = 8 waves (2x4).  Block tile 128x256, K-step 32, double-buffered
// LDS fed by GLOBAL_LOAD_ASYNC_TO_LDS_B128.  Each wave: 64x64 = 4x4 wmma tiles.
// ---------------------------------------------------------------------------
#define BM 128
#define BN 256
#define LDS_STRIDE 40   // 32 halves + 8 pad -> 80B rows, 16B aligned

__device__ __forceinline__ void async_ld16(uint32_t lds_addr, const half_t* gptr) {
    asm volatile("global_load_async_to_lds_b128 %0, %1, off"
                 :: "v"(lds_addr), "v"(gptr) : "memory");
}
__device__ __forceinline__ void wait_async0() {
    asm volatile("s_wait_asynccnt 0x0" ::: "memory");
}

__launch_bounds__(256)
__global__ void wmma_gemm(const half_t* __restrict__ A,
                          const half_t* __restrict__ Bt,
                          const float*  __restrict__ bias,
                          float* __restrict__ C,
                          int M, int N, int K) {
    __shared__ alignas(16) half_t As[2][BM * LDS_STRIDE];   // 2 x 10 KB
    __shared__ alignas(16) half_t Bs[2][BN * LDS_STRIDE];   // 2 x 20 KB

    const int tid  = threadIdx.x;
    const int lane = tid & 31;
    const int wave = tid >> 5;
    const int wm = wave >> 2;          // 0..1  (64-row group)
    const int wn = wave & 3;           // 0..3  (64-col group)
    const int lane15   = lane & 15;
    const int laneHalf = lane >> 4;    // 0 | 1

    const int rowBlock = blockIdx.y * BM;
    const int colBlock = blockIdx.x * BN;

    const v8f vzero = {0.f, 0.f, 0.f, 0.f, 0.f, 0.f, 0.f, 0.f};
    v8f acc[4][4];
#pragma unroll
    for (int im = 0; im < 4; ++im)
#pragma unroll
        for (int in = 0; in < 4; ++in) acc[im][in] = vzero;

    // stage one 128x32 A tile + 256x32 B tile into LDS buffer `buf`
    auto stage_tile = [&](int k0, int buf) {
#if STAGE_ASYNC
#pragma unroll
        for (int i = 0; i < 2; ++i) {                 // A: 512 16B chunks
            int c = tid + i * 256;
            int row = c >> 2, kc = (c & 3) * 8;
            async_ld16((uint32_t)(uintptr_t)&As[buf][row * LDS_STRIDE + kc],
                       A + (long)(rowBlock + row) * K + k0 + kc);
        }
#pragma unroll
        for (int i = 0; i < 4; ++i) {                 // B: 1024 16B chunks
            int c = tid + i * 256;
            int row = c >> 2, kc = (c & 3) * 8;
            async_ld16((uint32_t)(uintptr_t)&Bs[buf][row * LDS_STRIDE + kc],
                       Bt + (long)(colBlock + row) * K + k0 + kc);
        }
#else
        h8 ta[2], tb[4];
#pragma unroll
        for (int i = 0; i < 2; ++i) {
            int c = tid + i * 256;
            int row = c >> 2, kc = (c & 3) * 8;
            ta[i] = *(const h8*)(A + (long)(rowBlock + row) * K + k0 + kc);
        }
#pragma unroll
        for (int i = 0; i < 4; ++i) {
            int c = tid + i * 256;
            int row = c >> 2, kc = (c & 3) * 8;
            tb[i] = *(const h8*)(Bt + (long)(colBlock + row) * K + k0 + kc);
        }
#pragma unroll
        for (int i = 0; i < 2; ++i) {
            int c = tid + i * 256;
            int row = c >> 2, kc = (c & 3) * 8;
            *(h8*)(&As[buf][row * LDS_STRIDE + kc]) = ta[i];
        }
#pragma unroll
        for (int i = 0; i < 4; ++i) {
            int c = tid + i * 256;
            int row = c >> 2, kc = (c & 3) * 8;
            *(h8*)(&Bs[buf][row * LDS_STRIDE + kc]) = tb[i];
        }
#endif
    };

    const int nk = K >> 5;
    stage_tile(0, 0);

    for (int kt = 0; kt < nk; ++kt) {
        const int cur = kt & 1;
#if STAGE_ASYNC
        wait_async0();                 // my chunks of tile `kt` arrived in LDS
#endif
        __syncthreads();               // everyone's arrived; prev compute done
        if (kt + 1 < nk) stage_tile((kt + 1) << 5, cur ^ 1);

        // fragments (ISA 16-bit A 16x32 layout; analogous B)
        v16h afrag[4];
#pragma unroll
        for (int im = 0; im < 4; ++im) {
            int row = wm * 64 + im * 16 + lane15;
            int kh  = laneHalf * 8;
            union { v16h v; h8 h[2]; } u;
            u.h[0] = *(const h8*)(&As[cur][row * LDS_STRIDE + kh]);
            u.h[1] = *(const h8*)(&As[cur][row * LDS_STRIDE + kh + 16]);
            afrag[im] = u.v;
        }
        v16h bfrag[4];
#pragma unroll
        for (int in = 0; in < 4; ++in) {
            int col = wn * 64 + in * 16 + lane15;
            int kb  = laneHalf * 8;
            union { v16h v; h8 h[2]; } u;
            u.h[0] = *(const h8*)(&Bs[cur][col * LDS_STRIDE + kb]);
            u.h[1] = *(const h8*)(&Bs[cur][col * LDS_STRIDE + kb + 16]);
            bfrag[in] = u.v;
        }

#pragma unroll
        for (int im = 0; im < 4; ++im)
#pragma unroll
            for (int in = 0; in < 4; ++in)
                acc[im][in] = __builtin_amdgcn_wmma_f32_16x16x32_f16(
                    false, afrag[im], false, bfrag[in],
                    (short)0, acc[im][in], false, false);
    }

    // epilogue: C/D layout: VGPR j -> (M = j + 8*laneHalf, N = lane15)
#pragma unroll
    for (int im = 0; im < 4; ++im) {
        int rbase = rowBlock + wm * 64 + im * 16 + laneHalf * 8;
#pragma unroll
        for (int in = 0; in < 4; ++in) {
            int col = colBlock + wn * 64 + in * 16 + lane15;
            float bv = bias ? bias[col] : 0.0f;
#pragma unroll
            for (int j = 0; j < 8; ++j)
                C[(long)(rbase + j) * N + col] = acc[im][in][j] + bv;
        }
    }
}

// ---------------------------------------------------------------------------
// LSTM pointwise: gates [B,4096] (f|i|c|o), cell update, hidden update.
// ---------------------------------------------------------------------------
__global__ void lstm_pointwise(const float* __restrict__ gates,
                               const float* __restrict__ cell_in,
                               float* __restrict__ hidden_out,
                               float* __restrict__ cell_out,
                               half_t* __restrict__ Hbuf) {
    long idx = (long)blockIdx.x * blockDim.x + threadIdx.x;   // B*H threads
    long b = idx >> 10; int h = (int)(idx & 1023);
    const float* g = gates + b * NG;
    float f = g[h], i = g[1024 + h], c = g[2048 + h], o = g[3072 + h];
    float fg = 1.f / (1.f + __expf(-f));
    float ig = 1.f / (1.f + __expf(-i));
    float ct = tanhf(c);
    float og = 1.f / (1.f + __expf(-o));
    float cn = fg * cell_in[idx] + ig * ct;
    float hn = og * tanhf(cn);
    cell_out[idx]   = cn;
    hidden_out[idx] = hn;
    Hbuf[idx]       = (half_t)hn;
}

// ---------------------------------------------------------------------------
// Row-wise log_softmax over 1024 columns: one block per row.
// ---------------------------------------------------------------------------
__global__ void log_softmax_rows(const float* __restrict__ logits,
                                 float* __restrict__ out) {
    __shared__ float red[256];
    const int row = blockIdx.x;
    const float* x = logits + (long)row * H_DIM;
    float m = -INFINITY;
    for (int j = threadIdx.x; j < H_DIM; j += 256) m = fmaxf(m, x[j]);
    red[threadIdx.x] = m; __syncthreads();
    for (int s = 128; s > 0; s >>= 1) {
        if (threadIdx.x < s) red[threadIdx.x] = fmaxf(red[threadIdx.x], red[threadIdx.x + s]);
        __syncthreads();
    }
    m = red[0]; __syncthreads();
    float sum = 0.f;
    for (int j = threadIdx.x; j < H_DIM; j += 256) sum += __expf(x[j] - m);
    red[threadIdx.x] = sum; __syncthreads();
    for (int s = 128; s > 0; s >>= 1) {
        if (threadIdx.x < s) red[threadIdx.x] += red[threadIdx.x + s];
        __syncthreads();
    }
    float lse = m + __logf(red[0]);
    float* o = out + (long)row * H_DIM;
    for (int j = threadIdx.x; j < H_DIM; j += 256) o[j] = x[j] - lse;
}

// ---------------------------------------------------------------------------
// Launch
// ---------------------------------------------------------------------------
extern "C" void kernel_launch(void* const* d_in, const int* in_sizes, int n_in,
                              void* d_out, int out_size, void* d_ws, size_t ws_size,
                              hipStream_t stream) {
    const float* input  = (const float*)d_in[0];
    const float* hidden = (const float*)d_in[1];
    const float* cell   = (const float*)d_in[2];
    const float* W_f = (const float*)d_in[3];
    const float* U_f = (const float*)d_in[4];
    const float* b_f = (const float*)d_in[5];
    const float* W_i = (const float*)d_in[6];
    const float* U_i = (const float*)d_in[7];
    const float* b_i = (const float*)d_in[8];
    const float* W_c = (const float*)d_in[9];
    const float* U_c = (const float*)d_in[10];
    const float* b_c = (const float*)d_in[11];
    const float* W_o = (const float*)d_in[12];
    const float* U_o = (const float*)d_in[13];
    const float* b_o = (const float*)d_in[14];
    const float* W_h = (const float*)d_in[15];
    const float* b_h = (const float*)d_in[16];

    float* out = (float*)d_out;
    float* out_softmax = out;                 // [B,H]
    float* out_hidden  = out + (long)BHE;     // [B,H]
    float* out_cell    = out + 2L * BHE;      // [B,H]

    // ws layout (bytes, 256-aligned)
    char* ws = (char*)d_ws;
    half_t* Abuf   = (half_t*)(ws);                                   // 32 MB
    half_t* WUT    = (half_t*)(ws + 33554432L);                       // 16 MB
    half_t* WhT    = (half_t*)(ws + 50331648L);                       //  2 MB
    float*  bcat   = (float*) (ws + 52428800L);                       // 16 KB
    float*  gates  = (float*) (ws + 52445184L);                       // 128 MB
    half_t* Hbuf   = (half_t*)(ws + 186662912L);                      // 16 MB
    float*  logits = (float*) (ws + 203440128L);                      // 32 MB

    pack_A  <<<(B_DIM * KCAT) / 256, 256, 0, stream>>>(input, hidden, Abuf);
    pack_WUT<<<(NG * KCAT) / 256,   256, 0, stream>>>(W_f, W_i, W_c, W_o,
                                                      U_f, U_i, U_c, U_o, WUT);
    pack_bcat<<<NG / 256, 256, 0, stream>>>(b_f, b_i, b_c, b_o, bcat);
    pack_WhT<<<(H_DIM * H_DIM) / 256, 256, 0, stream>>>(W_h, WhT);

    // gates = [A|H] @ [W;U] + b        M=8192 N=4096 K=2048
    dim3 g1(NG / BN, B_DIM / BM);
    wmma_gemm<<<g1, 256, 0, stream>>>(Abuf, WUT, bcat, gates, B_DIM, NG, KCAT);

    lstm_pointwise<<<BHE / 256, 256, 0, stream>>>(gates, cell, out_hidden, out_cell, Hbuf);

    // logits = h_new @ W_h + b_h       M=8192 N=1024 K=1024
    dim3 g2(H_DIM / BN, B_DIM / BM);
    wmma_gemm<<<g2, 256, 0, stream>>>(Hbuf, WhT, b_h, logits, B_DIM, H_DIM, H_DIM);

    log_softmax_rows<<<B_DIM, 256, 0, stream>>>(logits, out_softmax);
}
